// SNNAlif_20675972563275
// MI455X (gfx1250) — matrix operand order repaired and checked by reading, not compiled, not run on hardware
//
#include <hip/hip_runtime.h>
#include <hip/hip_bf16.h>

// ---------------------------------------------------------------------------
// LSNN (adaptive LIF) forward for MI455X / gfx1250.
//   T=2000 B=64 F=128 H=512 O=64
// Critical path = 2000 sequential [64,512]x[512,512] recurrent GEMMs.
// Strategy: persistent 8-WGP kernel, weights + spike vector resident in the
// 320KB WGP LDS, neuron state in WMMA C-fragment registers, bf16 WMMA
// (spikes are exactly representable in bf16; accumulate f32).  The per-step
// spike-vector exchange (L2 -> LDS, 64KB contiguous) is done by the Tensor
// Data Mover (tensor_load_to_lds, TENSORcnt) instead of a VGPR round-trip.
// ---------------------------------------------------------------------------

typedef __attribute__((ext_vector_type(16))) __bf16       v16bf;
typedef __attribute__((ext_vector_type(8)))  float        v8f;
typedef __attribute__((ext_vector_type(4)))  unsigned int u32x4;
typedef __attribute__((ext_vector_type(4)))  int          i32x4;
typedef __attribute__((ext_vector_type(8)))  int          i32x8;

#define T_STEPS 2000
#define BATCH   64
#define FDIM    128
#define HDIM    512
#define ODIM    64
#define NWG     8              // persistent workgroups (one WGP each)
#define HSLICE  (HDIM / NWG)   // 64 output columns per WGP

// LSNN constants (DT=1e-3)
#define C_VDEC_V 0.9f                        // 1 - DT*TAU_MEM_INV
#define C_VDEC_I 0.1f                        // DT*TAU_MEM_INV
#define C_IDEC   0.8f                        // 1 - DT*TAU_SYN_INV
#define C_BDEC   1.4285714285714286e-6f      // DT*TAU_ADAPT_INV
#define C_BJUMP  2.5714285714285714e-3f      // TAU_ADAPT_INV*BETA
#define C_FILT   0.2231435511314f            // DT*TAU_FILTER_INV

// Tensor Data Mover availability / arity (see CDNA5_HIP.md: ROCm7.2 = 5-arg,
// therock/amdgpu-toolchain ships the TDM header and uses the 6-arg form).
#if defined(__has_builtin)
#  if __has_builtin(__builtin_amdgcn_tensor_load_to_lds)
#    define HAVE_TDM 1
#  endif
#endif
#ifndef HAVE_TDM
#  define HAVE_TDM 0
#endif
#if __has_include(<hip/amd_detail/amd_gfx1250_TDM.h>)
#  define TDM_SIX_ARGS 1
#else
#  define TDM_SIX_ARGS 0
#endif

// --- WMMA fragment loaders (ISA 7.12.2, 16-bit dense layouts) ---------------
// A (16x32, MxK): lane L holds row M=L&15; lanes<16: K=kb..kb+7 and kb+16..kb+23
// with kb=0; lanes>=16: same with kb=8.  Two 16B reads.
__device__ __forceinline__ v16bf frag_a(const __bf16* p) {
    v16bf r;
    ((u32x4*)&r)[0] = *(const u32x4*)(p);        // K = kb+0..7
    ((u32x4*)&r)[1] = *(const u32x4*)(p + 16);   // K = kb+16..23
    return r;
}
// B (32x16, KxN): lane L holds column N=L&15; lanes<16: K=0..7 and 16..23,
// lanes>=16: K=8..15 and 24..31 (same order as A).  Two 16B reads.
__device__ __forceinline__ v16bf frag_b(const __bf16* p) {
    v16bf r;
    ((u32x4*)&r)[0] = *(const u32x4*)(p);
    ((u32x4*)&r)[1] = *(const u32x4*)(p + 16);
    return r;
}

__device__ __forceinline__ v8f wmma_bf16(v16bf a, v16bf b, v8f c) {
    return __builtin_amdgcn_wmma_f32_16x16x32_bf16(false, a, false, b,
                                                   (short)0, c, false, false);
}

// --- TDM: contiguous 64KB (32768 x bf16) tile, global -> LDS ----------------
// D# layout per CDNA5 ISA: 1-D tensor,
// tensor_dim0 = tile_dim0 = 32768 elements, data_size = 2 bytes.
__device__ __forceinline__ void tdm_load_z(const __bf16* gsrc, unsigned lds_off)
{
#if HAVE_TDM
    const unsigned long long ga = (unsigned long long)(uintptr_t)gsrc;
    u32x4 g0;
    g0[0] = 1u;                                        // count=1 (user D#)
    g0[1] = lds_off;                                   // lds_addr (bytes)
    g0[2] = (unsigned)(ga & 0xFFFFFFFFu);              // global_addr[31:0]
    g0[3] = (unsigned)((ga >> 32) & 0x01FFFFFFu)       // global_addr[56:32]
          | (2u << 30);                                // type=2 ("image")
    i32x8 g1;
    g1[0] = (int)(1u << 16);        // wg_mask=0, data_size=1 (2B)
    g1[1] = (int)0x80000000u;       // tensor_dim0[15:0]=32768 in bits[63:48]
    g1[2] = (int)(1u << 16);        // tensor_dim0[31:16]=0, tensor_dim1[15:0]=1
    g1[3] = (int)0x80000000u;       // tensor_dim1[31:16]=0, tile_dim0=32768
    g1[4] = 1;                      // tile_dim1=1, tile_dim2=0
    g1[5] = 32768;                  // tensor_dim0_stride[31:0]
    g1[6] = 0;
    g1[7] = 0;
    i32x4 gz = {0, 0, 0, 0};
#if TDM_SIX_ARGS
    i32x8 gz8 = {0, 0, 0, 0, 0, 0, 0, 0};
    __builtin_amdgcn_tensor_load_to_lds(g0, g1, gz, gz, gz8, 0);
#else
    __builtin_amdgcn_tensor_load_to_lds(g0, g1, gz, gz, 0);
#endif
#else
    (void)gsrc; (void)lds_off;
#endif
}

// ---------------------------------------------------------------------------
__global__ void init_ws_kernel(unsigned* bar) { *bar = 0u; }

// ---------------------------------------------------------------------------
// Persistent scan kernel: grid = NWG blocks x 256 threads (8 waves).
// Each wave owns two 16x16 (b,h) output tiles of this WGP's [64 x 64] slice.
// ---------------------------------------------------------------------------
__launch_bounds__(256, 1)
__global__ void lsnn_scan_kernel(const float* __restrict__ x,
                                 const float* __restrict__ w_in,
                                 const float* __restrict__ w_rec,
                                 __bf16* __restrict__ spikes,   // [T,B,H] bf16
                                 float*  __restrict__ out_base, // d_out
                                 unsigned* __restrict__ bar)
{
    extern __shared__ char smem[];
    __bf16* wrec_s = (__bf16*)smem;               // [HSLICE][HDIM]  64 KB
    __bf16* win_s  = wrec_s + HSLICE * HDIM;      // [HSLICE][FDIM]  16 KB
    __bf16* z_s    = win_s  + HSLICE * FDIM;      // [BATCH][HDIM]   64 KB
    __bf16* x_s    = z_s    + BATCH  * HDIM;      // [BATCH][FDIM]   16 KB
    const unsigned z_lds_off =
        (unsigned)((HSLICE * HDIM + HSLICE * FDIM) * sizeof(__bf16)); // 81920

    const int tid  = threadIdx.x;
    const int lane = tid & 31;
    const int wave = tid >> 5;
    const int h0   = blockIdx.x * HSLICE;

    // Stage persistent weight slices (f32 -> bf16) once.
    for (int i = tid; i < HSLICE * FDIM; i += 256)
        win_s[i] = (__bf16)w_in[h0 * FDIM + i];
    for (int i = tid; i < HSLICE * HDIM; i += 256)
        wrec_s[i] = (__bf16)w_rec[h0 * HDIM + i];
    for (int i = tid; i < BATCH * HDIM; i += 256)
        z_s[i] = (__bf16)0.0f;                    // z_{-1} = 0

    // Neuron state held in WMMA C/D fragment layout:
    //   elem r, lane L -> b = m0 + r + 8*(L>>4), h = h0 + n0 + (L&15)
    v8f vv[2], ii[2], bb[2];
    float zz[2][8];
    for (int it = 0; it < 2; ++it)
        for (int r = 0; r < 8; ++r) { vv[it][r] = 0.f; ii[it][r] = 0.f;
                                      bb[it][r] = 0.f; zz[it][r] = 0.f; }

    const int ka = (lane >> 4) << 3;   // A k-base: 0 / 8
    const int kb = (lane >> 4) << 3;   // B k-base: 0 / 8

    for (int t = 0; t < T_STEPS; ++t) {
        __syncthreads();  // previous step's LDS consumers are done

        // Kick the TDM first so the DMA overlaps the x_t conversion below.
        if (HAVE_TDM && t > 0 && wave == 0)
            tdm_load_z(spikes + (size_t)(t - 1) * BATCH * HDIM, z_lds_off);

        // Stage x[t] (full batch) as bf16.
        const float* xt = x + (size_t)t * BATCH * FDIM;
        for (int i = tid; i < BATCH * FDIM; i += 256)
            x_s[i] = (__bf16)xt[i];

        if (!HAVE_TDM && t > 0) {   // fallback: VGPR round-trip copy
            const u32x4* src = (const u32x4*)(spikes + (size_t)(t - 1) * BATCH * HDIM);
            u32x4* dst = (u32x4*)z_s;
            for (int i = tid; i < (BATCH * HDIM) / 8; i += 256)
                dst[i] = src[i];
        }
        if (t + 1 < T_STEPS)   // warm L2/WGP$ for the next timestep's input
            __builtin_prefetch(x + (size_t)(t + 1) * BATCH * FDIM + tid * 32, 0, 0);

        if (HAVE_TDM && t > 0 && wave == 0)
            __builtin_amdgcn_s_wait_tensorcnt(0);   // DMA done before barrier
        __syncthreads();

        for (int it = 0; it < 2; ++it) {
            const int tile = wave * 2 + it;
            const int m0 = (tile >> 2) * 16;      // batch tile
            const int n0 = (tile & 3)  * 16;      // h tile (within slice)
            const int arow = m0 + (lane & 15);
            const int brow = n0 + (lane & 15);

            v8f acc; for (int r = 0; r < 8; ++r) acc[r] = 0.f;

            // input projection: K = FDIM = 128
            #pragma unroll
            for (int kt = 0; kt < FDIM / 32; ++kt) {
                v16bf a = frag_a(x_s  + arow * FDIM + kt * 32 + ka);
                v16bf b = frag_b(win_s + brow * FDIM + kt * 32 + kb);
                acc = wmma_bf16(a, b, acc);
            }
            // recurrence: K = HDIM = 512 (z_prev is exact in bf16)
            #pragma unroll 4
            for (int kt = 0; kt < HDIM / 32; ++kt) {
                v16bf a = frag_a(z_s   + arow * HDIM + kt * 32 + ka);
                v16bf b = frag_b(wrec_s + brow * HDIM + kt * 32 + kb);
                acc = wmma_bf16(a, b, acc);
            }

            // Pointwise ALIF update directly on the fragment.
            #pragma unroll
            for (int r = 0; r < 8; ++r) {
                float i_old = ii[it][r], v_old = vv[it][r], b_old = bb[it][r];
                float v_dec = C_VDEC_V * v_old + C_VDEC_I * i_old;
                float b_dec = b_old + C_BDEC * (1.0f - b_old);
                float z     = (v_dec - b_dec) > 0.0f ? 1.0f : 0.0f;
                vv[it][r] = (1.0f - z) * v_dec;
                ii[it][r] = C_IDEC * i_old + acc[r];   // i_dec + x_t·Wᵀ + z·Wrecᵀ
                bb[it][r] = b_dec + z * C_BJUMP;
                zz[it][r] = z;
                const int bg = m0 + r + ((lane >> 4) << 3);
                const int hg = h0 + n0 + (lane & 15);
                spikes[(size_t)t * BATCH * HDIM + (size_t)bg * HDIM + hg] = (__bf16)z;
            }
        }

        // Device-wide barrier: every WGP must see spikes[t] before step t+1.
        if (t + 1 < T_STEPS) {
            __threadfence();          // publish this thread's spike stores
            __syncthreads();
            if (tid == 0) {
                __hip_atomic_fetch_add(bar, 1u, __ATOMIC_RELEASE,
                                       __HIP_MEMORY_SCOPE_AGENT);
                const unsigned target = (unsigned)(NWG * (t + 1));
                while (__hip_atomic_load(bar, __ATOMIC_ACQUIRE,
                                         __HIP_MEMORY_SCOPE_AGENT) < target)
                    __builtin_amdgcn_s_sleep(2);
            }
            __syncthreads();
            __threadfence();          // acquire for all lanes
        }
    }

    // Final carry state (z,v,i,b) in f32 after the readout region of d_out.
    float* zT = out_base + (size_t)T_STEPS * BATCH * ODIM;
    float* vT = zT + BATCH * HDIM;
    float* iT = vT + BATCH * HDIM;
    float* bT = iT + BATCH * HDIM;
    for (int it = 0; it < 2; ++it) {
        const int tile = wave * 2 + it;
        const int m0 = (tile >> 2) * 16;
        const int n0 = (tile & 3)  * 16;
        for (int r = 0; r < 8; ++r) {
            const int b = m0 + r + ((lane >> 4) << 3);
            const int h = h0 + n0 + (lane & 15);
            const size_t idx = (size_t)b * HDIM + h;
            zT[idx] = zz[it][r];
            vT[idx] = vv[it][r];
            iT[idx] = ii[it][r];
            bT[idx] = bb[it][r];
        }
    }
}

// ---------------------------------------------------------------------------
// Readout GEMM: lin[m, o] = spikes[m, :] @ w_out[o, :] + b_out[o]
// M = T*B = 128000, N = 64, K = 512.  A comes straight from the bf16 spike
// history; w_out staged in LDS as bf16.  Writes f32 into d_out (filtered
// in place afterwards).
// ---------------------------------------------------------------------------
__launch_bounds__(256, 2)
__global__ void readout_kernel(const __bf16* __restrict__ spikes,
                               const float*  __restrict__ w_out,
                               const float*  __restrict__ b_out,
                               float* __restrict__ out)
{
    extern __shared__ char smem[];
    __bf16* wout_s = (__bf16*)smem;               // [ODIM][HDIM] 64 KB
    const int tid = threadIdx.x;
    for (int i = tid; i < ODIM * HDIM; i += 256)
        wout_s[i] = (__bf16)w_out[i];
    __syncthreads();

    const int lane = tid & 31, wave = tid >> 5;
    const int mBase = blockIdx.x * 32 + (wave >> 2) * 16;  // row in [T*B]
    const int n0    = (wave & 3) * 16;
    const int arow  = mBase + (lane & 15);
    const int brow  = n0 + (lane & 15);
    const int ka = (lane >> 4) << 3;
    const int kb = (lane >> 4) << 3;

    v8f acc; for (int r = 0; r < 8; ++r) acc[r] = 0.f;
    const __bf16* ap = spikes + (size_t)arow * HDIM;
    #pragma unroll 4
    for (int kt = 0; kt < HDIM / 32; ++kt) {
        v16bf a = frag_a(ap + kt * 32 + ka);
        v16bf b = frag_b(wout_s + brow * HDIM + kt * 32 + kb);
        acc = wmma_bf16(a, b, acc);
    }

    const float bias = b_out[n0 + (lane & 15)];
    for (int r = 0; r < 8; ++r) {
        const int row = mBase + r + ((lane >> 4) << 3);
        out[(size_t)row * ODIM + n0 + (lane & 15)] = acc[r] + bias;
    }
}

// ---------------------------------------------------------------------------
// Exponential low-pass over T, in place on d_out.  4096 independent lanes.
// ---------------------------------------------------------------------------
__global__ void filter_kernel(float* __restrict__ out)
{
    const int j = blockIdx.x * blockDim.x + threadIdx.x;   // b*O + o
    float y = 0.f;
    for (int t = 0; t < T_STEPS; ++t) {
        const size_t idx = (size_t)t * BATCH * ODIM + j;
        const float l = out[idx];
        y += C_FILT * (l - y);
        out[idx] = y;
    }
}

// ---------------------------------------------------------------------------
extern "C" void kernel_launch(void* const* d_in, const int* in_sizes, int n_in,
                              void* d_out, int out_size, void* d_ws, size_t ws_size,
                              hipStream_t stream)
{
    (void)in_sizes; (void)n_in; (void)out_size; (void)ws_size;
    const float* x     = (const float*)d_in[0];   // [T,B,F]
    const float* w_in  = (const float*)d_in[1];   // [H,F]
    const float* w_rec = (const float*)d_in[2];   // [H,H]
    const float* w_out = (const float*)d_in[3];   // [O,H]
    const float* b_out = (const float*)d_in[4];   // [O]
    float* out = (float*)d_out;

    // Workspace: bf16 spike history [T,B,H] (131 MB, L2-resident) + barrier.
    char* ws = (char*)d_ws;
    __bf16* spikes = (__bf16*)ws;
    const size_t spikes_bytes = (size_t)T_STEPS * BATCH * HDIM * sizeof(__bf16);
    unsigned* bar = (unsigned*)(ws + ((spikes_bytes + 255) & ~(size_t)255));

    init_ws_kernel<<<1, 1, 0, stream>>>(bar);

    const size_t scan_lds = (size_t)(HSLICE * HDIM + HSLICE * FDIM +
                                     BATCH * HDIM + BATCH * FDIM) * sizeof(__bf16);
    hipFuncSetAttribute(reinterpret_cast<const void*>(lsnn_scan_kernel),
                        hipFuncAttributeMaxDynamicSharedMemorySize, (int)scan_lds);
    lsnn_scan_kernel<<<NWG, 256, scan_lds, stream>>>(x, w_in, w_rec, spikes, out, bar);

    const size_t ro_lds = (size_t)ODIM * HDIM * sizeof(__bf16);
    hipFuncSetAttribute(reinterpret_cast<const void*>(readout_kernel),
                        hipFuncAttributeMaxDynamicSharedMemorySize, (int)ro_lds);
    readout_kernel<<<(T_STEPS * BATCH) / 32, 256, ro_lds, stream>>>(spikes, w_out, b_out, out);

    filter_kernel<<<(BATCH * ODIM) / 256, 256, 0, stream>>>(out);
}
